// HRMModel_48344151883864
// MI455X (gfx1250) — compile-verified, matrix operand order checked by reading, not run on hardware
//
#include <hip/hip_runtime.h>
#include <hip/hip_bf16.h>
#include <cstdint>
#include <cstddef>

// ---------------------------------------------------------------------------
// Types / constants
// ---------------------------------------------------------------------------
typedef __attribute__((ext_vector_type(16))) __bf16 v16bf;
typedef __attribute__((ext_vector_type(8)))  __bf16 v8bf;
typedef __attribute__((ext_vector_type(8)))  float  v8f;

#define BM      32          // batch rows per workgroup
#define BLOCK   256         // 8 waves (wave32)
#define IN_DIM  2048
#define HD      512
#define LD      256
#define LDH     520         // padded bf16 row stride (1040B -> 4-bank skew)
#define LDL     264         // padded bf16 row stride for zL
#define LTH     516         // padded f32 staging stride (H-sized)
#define LTL     260         // padded f32 staging stride (L-sized)
#define FD_EPS_F 1e-3f

__device__ __forceinline__ float bf2f(__bf16 v) { return (float)v; }
__device__ __forceinline__ __bf16 f2bf(float v) { return (__bf16)v; }
__device__ __forceinline__ float sigm(float x) { return 1.0f / (1.0f + __expf(-x)); }
__device__ __forceinline__ float softplus(float x) {
  return (x > 20.0f) ? x : log1pf(__expf(x));
}
__device__ __forceinline__ float red8(float v) {
  v += __shfl_xor(v, 1, 32); v += __shfl_xor(v, 2, 32); v += __shfl_xor(v, 4, 32);
  return v;
}
__device__ __forceinline__ float red32(float v) {
  v += __shfl_xor(v, 1, 32);  v += __shfl_xor(v, 2, 32);  v += __shfl_xor(v, 4, 32);
  v += __shfl_xor(v, 8, 32);  v += __shfl_xor(v, 16, 32);
  return v;
}

// D = A(16x32 bf16) x B(32x16 bf16) + C (16x16 f32)
__device__ __forceinline__ v8f wmma_bf16(v16bf a, v16bf b, v8f c) {
  return __builtin_amdgcn_wmma_f32_16x16x32_bf16(false, a, false, b, (short)0, c,
                                                 false, false);
}

// A-fragment from LDS, row-major [rows, ldk] bf16 (ldk padded).
__device__ __forceinline__ v16bf load_a(const __bf16* s, int ldk, int m0, int k0) {
  int lane = threadIdx.x & 31;
  const __bf16* p = s + (m0 + (lane & 15)) * ldk + k0 + ((lane >> 4) << 3);
  v8bf lo = *(const v8bf*)p;
  v8bf hi = *(const v8bf*)(p + 16);
  v16bf a;
#pragma unroll
  for (int j = 0; j < 8; ++j) { a[j] = lo[j]; a[j + 8] = hi[j]; }
  return a;
}

// B-fragment from pre-swizzled weights: tile t = kt*ntn + nt, lane-contiguous.
__device__ __forceinline__ v16bf load_b(const __bf16* w, int ntn, int kt, int nt) {
  int lane = threadIdx.x & 31;
  return *(const v16bf*)(w + ((size_t)(kt * ntn + nt) * 32 + lane) * 16);
}

// C/D tile (16x16 f32) -> LDS row-major [*, ldn]
__device__ __forceinline__ void store_c(float* s, int ldn, int m0, int n0, v8f c) {
  int lane = threadIdx.x & 31;
  float* p = s + (m0 + ((lane >> 4) << 3)) * ldn + n0 + (lane & 15);
#pragma unroll
  for (int r = 0; r < 8; ++r) p[r * ldn] = c[r];
}

// Async global->LDS tile stage (gfx1250): per-lane 16B chunks, ASYNCcnt tracked.
// gsrc: contiguous [rows, cols] bf16;  ldst: LDS with padded row stride ldl.
__device__ __forceinline__ void async_copy_rows(const __bf16* gsrc, __bf16* ldst,
                                                int rows, int cols, int ldl) {
  int cpr = cols >> 3;  // 16B chunks per row
  int total = rows * cpr;
  for (int c = threadIdx.x; c < total; c += BLOCK) {
    int r = c / cpr, k = c - r * cpr;
    unsigned lds = (unsigned)(size_t)(ldst + r * ldl + k * 8);
    const __bf16* g = gsrc + (size_t)r * cols + k * 8;
    asm volatile("global_load_async_to_lds_b128 %0, %1, off"
                 :: "v"(lds), "v"(g) : "memory");
  }
  asm volatile("s_wait_asynccnt 0x0" ::: "memory");
}

// GRU gate math on one 16x16 output tile (C-layout registers).
__device__ __forceinline__ void gru_gate(
    const v8f& gr, const v8f& gz, const v8f& gn,
    const v8f& hr, const v8f& hz, const v8f& hn,
    float bir, float biz, float bin_, float bhr, float bhz, float bhn,
    int mrow, int cc, const __bf16* s_state, int ldst,
    float* s_out, int ldout) {
#pragma unroll
  for (int r = 0; r < 8; ++r) {
    int m = mrow + r;
    float rr = sigm(gr[r] + bir + hr[r] + bhr);
    float zz = sigm(gz[r] + biz + hz[r] + bhz);
    float nn = tanhf(gn[r] + bin_ + rr * (hn[r] + bhn));
    float h = bf2f(s_state[m * ldst + cc]);
    s_out[m * ldout + cc] = (1.f - zz) * nn + zz * h;
  }
}

#define VZERO {0,0,0,0,0,0,0,0}

// ---------------------------------------------------------------------------
// Weight swizzle: fp32 [K,N] row-major -> bf16 WMMA-B fragment-major
// ---------------------------------------------------------------------------
__global__ void swizzle_w(const float* __restrict__ W, __bf16* __restrict__ out,
                          int K, int N) {
  int ntn = N >> 4, ntk = K >> 5;
  int total = ntk * ntn * 32;
  for (int i = blockIdx.x * blockDim.x + threadIdx.x; i < total;
       i += gridDim.x * blockDim.x) {
    int lane = i & 31;
    int t = i >> 5;
    int nt = t % ntn, kt = t / ntn;
    int n = nt * 16 + (lane & 15);
    int kbase = kt * 32 + ((lane >> 4) << 3);
    __bf16* dst = out + (size_t)i * 16;
#pragma unroll
    for (int j = 0; j < 8; ++j) {
      dst[j]     = f2bf(W[(size_t)(kbase + j) * N + n]);
      dst[j + 8] = f2bf(W[(size_t)(kbase + 16 + j) * N + n]);
    }
  }
}

// ---------------------------------------------------------------------------
// noise row inverse norms
// ---------------------------------------------------------------------------
__global__ void rownorm_kernel(const float* __restrict__ noise,
                               float* __restrict__ invn, int B) {
  int row = blockIdx.x * 8 + (threadIdx.x >> 5);
  int lane = threadIdx.x & 31;
  if (row >= B) return;
  float ss = 0.f;
  const float* p = noise + (size_t)row * IN_DIM;
  for (int j = lane; j < IN_DIM; j += 32) { float v = p[j]; ss += v * v; }
  ss = red32(ss);
  if (lane == 0) invn[row] = 1.0f / fmaxf(sqrtf(ss), 1e-12f);
}

// ---------------------------------------------------------------------------
// Projection: x_tilde = rmsnorm((x [+ eps*noise/||noise||]) @ proj_w + b, nw)
// ---------------------------------------------------------------------------
template <bool EPS>
__device__ __forceinline__ v16bf make_a_global(const float* __restrict__ px,
                                               const float* __restrict__ pn,
                                               float sc) {
  float4 x0 = *(const float4*)px,        x1 = *(const float4*)(px + 4);
  float4 x2 = *(const float4*)(px + 16), x3 = *(const float4*)(px + 20);
  if (EPS) {
    float4 n0 = *(const float4*)pn,        n1 = *(const float4*)(pn + 4);
    float4 n2 = *(const float4*)(pn + 16), n3 = *(const float4*)(pn + 20);
    x0.x += sc*n0.x; x0.y += sc*n0.y; x0.z += sc*n0.z; x0.w += sc*n0.w;
    x1.x += sc*n1.x; x1.y += sc*n1.y; x1.z += sc*n1.z; x1.w += sc*n1.w;
    x2.x += sc*n2.x; x2.y += sc*n2.y; x2.z += sc*n2.z; x2.w += sc*n2.w;
    x3.x += sc*n3.x; x3.y += sc*n3.y; x3.z += sc*n3.z; x3.w += sc*n3.w;
  }
  v16bf a;
  a[0]=f2bf(x0.x); a[1]=f2bf(x0.y); a[2]=f2bf(x0.z); a[3]=f2bf(x0.w);
  a[4]=f2bf(x1.x); a[5]=f2bf(x1.y); a[6]=f2bf(x1.z); a[7]=f2bf(x1.w);
  a[8]=f2bf(x2.x); a[9]=f2bf(x2.y); a[10]=f2bf(x2.z); a[11]=f2bf(x2.w);
  a[12]=f2bf(x3.x); a[13]=f2bf(x3.y); a[14]=f2bf(x3.z); a[15]=f2bf(x3.w);
  return a;
}

template <bool EPS>
__global__ __launch_bounds__(BLOCK) void proj_kernel(
    const float* __restrict__ x, const float* __restrict__ noise,
    const float* __restrict__ invn, const __bf16* __restrict__ w_proj,
    const float* __restrict__ proj_b, const float* __restrict__ proj_nw,
    __bf16* __restrict__ xt_out) {
  __shared__ float s_tmp[BM * LTH];
  int tid = threadIdx.x, wid = tid >> 5, lane = tid & 31;
  int row0 = blockIdx.x * BM;

  int m0 = row0 + (lane & 15);
  int m1 = m0 + 16;
  float sc0 = EPS ? FD_EPS_F * invn[m0] : 0.f;
  float sc1 = EPS ? FD_EPS_F * invn[m1] : 0.f;
  v8f a00 = VZERO, a01 = VZERO, a02 = VZERO, a03 = VZERO;
  v8f a10 = VZERO, a11 = VZERO, a12 = VZERO, a13 = VZERO;
  for (int kt = 0; kt < IN_DIM / 32; ++kt) {
    int kb = kt * 32 + ((lane >> 4) << 3);
    v16bf f0 = make_a_global<EPS>(x + (size_t)m0 * IN_DIM + kb,
                                  noise + (size_t)m0 * IN_DIM + kb, sc0);
    v16bf f1 = make_a_global<EPS>(x + (size_t)m1 * IN_DIM + kb,
                                  noise + (size_t)m1 * IN_DIM + kb, sc1);
    v16bf b0 = load_b(w_proj, HD / 16, kt, wid * 4 + 0);
    v16bf b1 = load_b(w_proj, HD / 16, kt, wid * 4 + 1);
    v16bf b2 = load_b(w_proj, HD / 16, kt, wid * 4 + 2);
    v16bf b3 = load_b(w_proj, HD / 16, kt, wid * 4 + 3);
    a00 = wmma_bf16(f0, b0, a00); a01 = wmma_bf16(f0, b1, a01);
    a02 = wmma_bf16(f0, b2, a02); a03 = wmma_bf16(f0, b3, a03);
    a10 = wmma_bf16(f1, b0, a10); a11 = wmma_bf16(f1, b1, a11);
    a12 = wmma_bf16(f1, b2, a12); a13 = wmma_bf16(f1, b3, a13);
  }
  store_c(s_tmp, LTH, 0,  (wid * 4 + 0) * 16, a00);
  store_c(s_tmp, LTH, 0,  (wid * 4 + 1) * 16, a01);
  store_c(s_tmp, LTH, 0,  (wid * 4 + 2) * 16, a02);
  store_c(s_tmp, LTH, 0,  (wid * 4 + 3) * 16, a03);
  store_c(s_tmp, LTH, 16, (wid * 4 + 0) * 16, a10);
  store_c(s_tmp, LTH, 16, (wid * 4 + 1) * 16, a11);
  store_c(s_tmp, LTH, 16, (wid * 4 + 2) * 16, a12);
  store_c(s_tmp, LTH, 16, (wid * 4 + 3) * 16, a13);
  __syncthreads();
  // bias + rmsnorm (8 threads/row)
  {
    int row = tid >> 3, l8 = tid & 7;
    size_t grow = (size_t)row0 + row;
    float ss = 0.f;
    for (int j = 0; j < 64; ++j) {
      int e = l8 + 8 * j;
      float v = s_tmp[row * LTH + e] + proj_b[e];
      ss += v * v;
    }
    ss = red8(ss);
    float sc = rsqrtf(ss / (float)HD + 1e-6f);
    for (int j = 0; j < 64; ++j) {
      int e = l8 + 8 * j;
      float v = s_tmp[row * LTH + e] + proj_b[e];
      xt_out[grow * HD + e] = f2bf(v * sc * proj_nw[e]);
    }
  }
}

// ---------------------------------------------------------------------------
// Rollout: 4 cycles x (4 L-steps + 1 H-step), state resident in LDS.
// ---------------------------------------------------------------------------
__global__ __launch_bounds__(BLOCK) void rollout_kernel(
    const __bf16* __restrict__ xt_all,
    const __bf16* __restrict__ w_lih, const __bf16* __restrict__ w_lhh,
    const __bf16* __restrict__ w_hih, const __bf16* __restrict__ w_hhh,
    const float* __restrict__ lb_ih, const float* __restrict__ lb_hh,
    const float* __restrict__ l_nw,
    const float* __restrict__ hb_ih, const float* __restrict__ hb_hh,
    const float* __restrict__ h_nw, const float* __restrict__ fin_nw,
    __bf16* __restrict__ zh_head_out,
    float* __restrict__ out_zL, float* __restrict__ out_zH,
    float* __restrict__ out_halt, int write_full) {
  __shared__ __bf16 s_xt[BM * LDH];
  __shared__ __bf16 s_zh[BM * LDH];
  __shared__ __bf16 s_zl[BM * LDL];
  __shared__ __bf16 s_zmax[BM * HD];
  __shared__ float  s_tmp[BM * LTH];

  int tid = threadIdx.x, wid = tid >> 5, lane = tid & 31;
  int row0 = blockIdx.x * BM;

  // async-stage x_tilde tile; zero states
  async_copy_rows(xt_all + (size_t)row0 * HD, s_xt, BM, HD, LDH);
  for (int i = tid; i < BM * LDH; i += BLOCK) s_zh[i] = f2bf(0.f);
  for (int i = tid; i < BM * HD; i += BLOCK) s_zmax[i] = f2bf(0.f);
  for (int i = tid; i < BM * LDL; i += BLOCK) s_zl[i] = f2bf(0.f);
  __syncthreads();

  for (int cyc = 0; cyc < 4; ++cyc) {
    // ---------------- L-steps ----------------
    for (int t = 0; t < 4; ++t) {
      for (int q = 0; q < 2; ++q) {
        int cg = wid * 2 + q;  // gate col-group 0..15
        v8f gr0 = VZERO, gz0 = VZERO, gn0 = VZERO;
        v8f gr1 = VZERO, gz1 = VZERO, gn1 = VZERO;
        for (int kt = 0; kt < 16; ++kt) {  // x_tilde part of l_in
          v16bf br = load_b(w_lih, 48, kt, cg);
          v16bf bz = load_b(w_lih, 48, kt, cg + 16);
          v16bf bn = load_b(w_lih, 48, kt, cg + 32);
          v16bf a0 = load_a(s_xt, LDH, 0, kt * 32);
          v16bf a1 = load_a(s_xt, LDH, 16, kt * 32);
          gr0 = wmma_bf16(a0, br, gr0); gz0 = wmma_bf16(a0, bz, gz0);
          gn0 = wmma_bf16(a0, bn, gn0);
          gr1 = wmma_bf16(a1, br, gr1); gz1 = wmma_bf16(a1, bz, gz1);
          gn1 = wmma_bf16(a1, bn, gn1);
        }
        for (int kt = 0; kt < 16; ++kt) {  // zH part of l_in
          v16bf br = load_b(w_lih, 48, 16 + kt, cg);
          v16bf bz = load_b(w_lih, 48, 16 + kt, cg + 16);
          v16bf bn = load_b(w_lih, 48, 16 + kt, cg + 32);
          v16bf a0 = load_a(s_zh, LDH, 0, kt * 32);
          v16bf a1 = load_a(s_zh, LDH, 16, kt * 32);
          gr0 = wmma_bf16(a0, br, gr0); gz0 = wmma_bf16(a0, bz, gz0);
          gn0 = wmma_bf16(a0, bn, gn0);
          gr1 = wmma_bf16(a1, br, gr1); gz1 = wmma_bf16(a1, bz, gz1);
          gn1 = wmma_bf16(a1, bn, gn1);
        }
        v8f hr0 = VZERO, hz0 = VZERO, hn0 = VZERO;
        v8f hr1 = VZERO, hz1 = VZERO, hn1 = VZERO;
        for (int kt = 0; kt < 8; ++kt) {   // zL @ l_w_hh
          v16bf br = load_b(w_lhh, 48, kt, cg);
          v16bf bz = load_b(w_lhh, 48, kt, cg + 16);
          v16bf bn = load_b(w_lhh, 48, kt, cg + 32);
          v16bf a0 = load_a(s_zl, LDL, 0, kt * 32);
          v16bf a1 = load_a(s_zl, LDL, 16, kt * 32);
          hr0 = wmma_bf16(a0, br, hr0); hz0 = wmma_bf16(a0, bz, hz0);
          hn0 = wmma_bf16(a0, bn, hn0);
          hr1 = wmma_bf16(a1, br, hr1); hz1 = wmma_bf16(a1, bz, hz1);
          hn1 = wmma_bf16(a1, bn, hn1);
        }
        int cc = cg * 16 + (lane & 15);  // 0..255
        float bir = lb_ih[cc], biz = lb_ih[LD + cc], bin_ = lb_ih[2 * LD + cc];
        float bhr = lb_hh[cc], bhz = lb_hh[LD + cc], bhn = lb_hh[2 * LD + cc];
        int mr = (lane >> 4) << 3;
        gru_gate(gr0, gz0, gn0, hr0, hz0, hn0, bir, biz, bin_, bhr, bhz, bhn,
                 mr, cc, s_zl, LDL, s_tmp, LTL);
        gru_gate(gr1, gz1, gn1, hr1, hz1, hn1, bir, biz, bin_, bhr, bhz, bhn,
                 16 + mr, cc, s_zl, LDL, s_tmp, LTL);
      }
      __syncthreads();
      { // rmsnorm -> s_zl
        int row = tid >> 3, l8 = tid & 7;
        float ss = 0.f;
        for (int j = 0; j < 32; ++j) {
          float v = s_tmp[row * LTL + l8 + 8 * j];
          ss += v * v;
        }
        ss = red8(ss);
        float sc = rsqrtf(ss / (float)LD + 1e-6f);
        for (int j = 0; j < 32; ++j) {
          int e = l8 + 8 * j;
          s_zl[row * LDL + e] = f2bf(s_tmp[row * LTL + e] * sc * l_nw[e]);
        }
      }
      __syncthreads();
    }

    // ---------------- H-step ----------------
    for (int q = 0; q < 4; ++q) {
      int cg = wid * 4 + q;  // gate col-group 0..31
      v8f gr0 = VZERO, gz0 = VZERO, gn0 = VZERO;
      v8f gr1 = VZERO, gz1 = VZERO, gn1 = VZERO;
      for (int kt = 0; kt < 8; ++kt) {   // zL part of h_in
        v16bf br = load_b(w_hih, 96, kt, cg);
        v16bf bz = load_b(w_hih, 96, kt, cg + 32);
        v16bf bn = load_b(w_hih, 96, kt, cg + 64);
        v16bf a0 = load_a(s_zl, LDL, 0, kt * 32);
        v16bf a1 = load_a(s_zl, LDL, 16, kt * 32);
        gr0 = wmma_bf16(a0, br, gr0); gz0 = wmma_bf16(a0, bz, gz0);
        gn0 = wmma_bf16(a0, bn, gn0);
        gr1 = wmma_bf16(a1, br, gr1); gz1 = wmma_bf16(a1, bz, gz1);
        gn1 = wmma_bf16(a1, bn, gn1);
      }
      for (int kt = 0; kt < 16; ++kt) {  // zH part of h_in
        v16bf br = load_b(w_hih, 96, 8 + kt, cg);
        v16bf bz = load_b(w_hih, 96, 8 + kt, cg + 32);
        v16bf bn = load_b(w_hih, 96, 8 + kt, cg + 64);
        v16bf a0 = load_a(s_zh, LDH, 0, kt * 32);
        v16bf a1 = load_a(s_zh, LDH, 16, kt * 32);
        gr0 = wmma_bf16(a0, br, gr0); gz0 = wmma_bf16(a0, bz, gz0);
        gn0 = wmma_bf16(a0, bn, gn0);
        gr1 = wmma_bf16(a1, br, gr1); gz1 = wmma_bf16(a1, bz, gz1);
        gn1 = wmma_bf16(a1, bn, gn1);
      }
      v8f hr0 = VZERO, hz0 = VZERO, hn0 = VZERO;
      v8f hr1 = VZERO, hz1 = VZERO, hn1 = VZERO;
      for (int kt = 0; kt < 16; ++kt) {  // zH @ h_w_hh
        v16bf br = load_b(w_hhh, 96, kt, cg);
        v16bf bz = load_b(w_hhh, 96, kt, cg + 32);
        v16bf bn = load_b(w_hhh, 96, kt, cg + 64);
        v16bf a0 = load_a(s_zh, LDH, 0, kt * 32);
        v16bf a1 = load_a(s_zh, LDH, 16, kt * 32);
        hr0 = wmma_bf16(a0, br, hr0); hz0 = wmma_bf16(a0, bz, hz0);
        hn0 = wmma_bf16(a0, bn, hn0);
        hr1 = wmma_bf16(a1, br, hr1); hz1 = wmma_bf16(a1, bz, hz1);
        hn1 = wmma_bf16(a1, bn, hn1);
      }
      int cc = cg * 16 + (lane & 15);  // 0..511
      float bir = hb_ih[cc], biz = hb_ih[HD + cc], bin_ = hb_ih[2 * HD + cc];
      float bhr = hb_hh[cc], bhz = hb_hh[HD + cc], bhn = hb_hh[2 * HD + cc];
      int mr = (lane >> 4) << 3;
      gru_gate(gr0, gz0, gn0, hr0, hz0, hn0, bir, biz, bin_, bhr, bhz, bhn,
               mr, cc, s_zh, LDH, s_tmp, LTH);
      gru_gate(gr1, gz1, gn1, hr1, hz1, hn1, bir, biz, bin_, bhr, bhz, bhn,
               16 + mr, cc, s_zh, LDH, s_tmp, LTH);
    }
    __syncthreads();
    { // rmsnorm -> s_zh, update s_zmax
      int row = tid >> 3, l8 = tid & 7;
      float ss = 0.f;
      for (int j = 0; j < 64; ++j) {
        float v = s_tmp[row * LTH + l8 + 8 * j];
        ss += v * v;
      }
      ss = red8(ss);
      float sc = rsqrtf(ss / (float)HD + 1e-6f);
      for (int j = 0; j < 64; ++j) {
        int e = l8 + 8 * j;
        float y = s_tmp[row * LTH + e] * sc * h_nw[e];
        s_zh[row * LDH + e] = f2bf(y);
        float zm = bf2f(s_zmax[row * HD + e]);
        s_zmax[row * HD + e] = f2bf(fmaxf(zm, y));
      }
    }
    __syncthreads();
  }

  // ---------------- final norms / outputs ----------------
  {
    int row = tid >> 3, l8 = tid & 7;
    size_t grow = (size_t)row0 + row;
    float ss = 0.f;
    for (int j = 0; j < 64; ++j) {
      float v = bf2f(s_zh[row * LDH + l8 + 8 * j]);
      ss += v * v;
    }
    ss = red8(ss);
    float sc = rsqrtf(ss / (float)HD + 1e-6f);
    for (int j = 0; j < 64; ++j) {
      int e = l8 + 8 * j;
      float y = bf2f(s_zh[row * LDH + e]) * sc * fin_nw[e];
      zh_head_out[grow * HD + e] = f2bf(y);
      if (write_full) {
        out_zH[grow * HD + e] = y;
        s_tmp[row * LTH + e] = y;
      }
    }
    if (write_full) {
      __syncthreads();
      float sm = 0.f;
      for (int j = 0; j < 64; ++j) {
        float v = bf2f(s_zmax[row * HD + l8 + 8 * j]);
        sm += v * v;
      }
      sm = red8(sm);
      float scm = rsqrtf(sm / (float)HD + 1e-6f);
      float dot = 0.f;
      for (int j = 0; j < 64; ++j) {
        int e = l8 + 8 * j;
        dot += bf2f(s_zmax[row * HD + e]) * scm * fin_nw[e] * s_tmp[row * LTH + e];
      }
      dot = red8(dot);
      if (l8 == 0)
        out_halt[grow] = sigm(dot * (1.0f / ((float)HD * (float)HD)));
      for (int i = tid; i < BM * LD; i += BLOCK) {
        int r = i >> 8, e = i & 255;
        out_zL[((size_t)row0 + r) * LD + e] = bf2f(s_zl[r * LDL + e]);
      }
    }
  }
}

// ---------------------------------------------------------------------------
// Heads: recon GEMM (WMMA) + all per-row scalar heads
// ---------------------------------------------------------------------------
__global__ __launch_bounds__(BLOCK) void heads_kernel(
    const __bf16* __restrict__ zhh, const __bf16* __restrict__ zhe,
    const __bf16* __restrict__ xt, const float* __restrict__ mask,
    const __bf16* __restrict__ w_rec, const float* __restrict__ recon_b,
    const float* __restrict__ score_w, const float* __restrict__ score_b,
    const float* __restrict__ temp_w, const float* __restrict__ temp_b,
    const float* __restrict__ logvar_w, const float* __restrict__ logvar_b,
    const float* __restrict__ aux3_w, const float* __restrict__ aux3_b,
    const float* __restrict__ dis_w, const float* __restrict__ dis_b,
    const float* __restrict__ cons_w, const float* __restrict__ cons_b,
    const float* __restrict__ ood_w, const float* __restrict__ ood_b,
    float* __restrict__ out, int B) {
  __shared__ __bf16 s_a[BM * LDH];
  __shared__ float  s_rec[BM * LTH];
  int tid = threadIdx.x, wid = tid >> 5, lane = tid & 31;
  int row0 = blockIdx.x * BM;
  size_t Bz = (size_t)B;

  const size_t o_score01 = 0, o_slogit = Bz, o_logvar = 2 * Bz, o_aux3 = 3 * Bz;
  const size_t o_dis = 6 * Bz, o_conshat = 7 * Bz, o_ctgt = 8 * Bz;
  const size_t o_xrec = 9 * Bz, o_rsim = 521 * Bz, o_ood = 522 * Bz;
  const size_t o_jac = 523 * Bz;

  async_copy_rows(zhh + (size_t)row0 * HD, s_a, BM, HD, LDH);
  __syncthreads();

  // x_recon = zH_head @ recon_w + recon_b  (both m-tiles share B-frags)
  for (int q = 0; q < 4; ++q) {
    int nt = wid * 4 + q;
    v8f acc0 = VZERO, acc1 = VZERO;
    for (int kt = 0; kt < 16; ++kt) {
      v16bf b = load_b(w_rec, 32, kt, nt);
      v16bf a0 = load_a(s_a, LDH, 0, kt * 32);
      v16bf a1 = load_a(s_a, LDH, 16, kt * 32);
      acc0 = wmma_bf16(a0, b, acc0);
      acc1 = wmma_bf16(a1, b, acc1);
    }
    int n = nt * 16 + (lane & 15);
    int mB = (lane >> 4) << 3;
    float bb = recon_b[n];
#pragma unroll
    for (int r = 0; r < 8; ++r) {
      float v0 = acc0[r] + bb;
      float v1 = acc1[r] + bb;
      s_rec[(mB + r) * LTH + n] = v0;
      s_rec[(16 + mB + r) * LTH + n] = v1;
      out[o_xrec + ((size_t)row0 + mB + r) * HD + n] = v0;
      out[o_xrec + ((size_t)row0 + 16 + mB + r) * HD + n] = v1;
    }
  }
  __syncthreads();

  // per-row heads (8 threads/row)
  {
    int row = tid >> 3, l8 = tid & 7;
    size_t grow = (size_t)row0 + row;
    float ds = 0, dt = 0, dl = 0, a0 = 0, a1 = 0, a2 = 0;
    float dd = 0, dc = 0, doo = 0, s2 = 0, s2m = 0;
    float rdot = 0, rn = 0, xn = 0, dse = 0, dte = 0;
    for (int j = 0; j < 64; ++j) {
      int e = l8 + 8 * j;
      float v  = bf2f(s_a[row * LDH + e]);
      float ve = bf2f(zhe[grow * HD + e]);
      float xv = bf2f(xt[grow * HD + e]);
      float rv = s_rec[row * LTH + e];
      ds += v * score_w[e];  dt += v * temp_w[e];  dl += v * logvar_w[e];
      a0 += v * aux3_w[e * 3 + 0]; a1 += v * aux3_w[e * 3 + 1];
      a2 += v * aux3_w[e * 3 + 2];
      dd += v * dis_w[e];  dc += v * cons_w[e];  doo += v * ood_w[e];
      float mk = (mask[grow * HD + e] < 0.1f) ? 1.f : 0.f;
      s2 += v * v;  s2m += v * v * (1.f - mk);
      rdot += rv * xv;  rn += rv * rv;  xn += xv * xv;
      dse += ve * score_w[e];  dte += ve * temp_w[e];
    }
    ds = red8(ds); dt = red8(dt); dl = red8(dl);
    a0 = red8(a0); a1 = red8(a1); a2 = red8(a2);
    dd = red8(dd); dc = red8(dc); doo = red8(doo);
    s2 = red8(s2); s2m = red8(s2m);
    rdot = red8(rdot); rn = red8(rn); xn = red8(xn);
    dse = red8(dse); dte = red8(dte);
    if (l8 == 0) {
      float slog = ds + score_b[0];
      float tau = 0.5f + 0.5f * softplus(dt + temp_b[0]);
      float sc01 = sigm(slog / tau);
      out[o_score01 + grow] = sc01;
      out[o_slogit + grow] = slog;
      out[o_logvar + grow] = dl + logvar_b[0];
      out[o_aux3 + grow * 3 + 0] = a0 + aux3_b[0];
      out[o_aux3 + grow * 3 + 1] = a1 + aux3_b[1];
      out[o_aux3 + grow * 3 + 2] = a2 + aux3_b[2];
      out[o_dis + grow] = sigm(dd + dis_b[0]);
      out[o_conshat + grow] = sigm(dc + cons_b[0]);
      out[o_ctgt + grow] =
          (s2m / fmaxf(sqrtf(s2) * sqrtf(s2m), 1e-6f) + 1.f) * 0.5f;
      out[o_rsim + grow] =
          (rdot / fmaxf(sqrtf(rn) * sqrtf(xn), 1e-6f) + 1.f) * 0.5f;
      out[o_ood + grow] = sigm(doo + ood_b[0]);
      float taue = 0.5f + 0.5f * softplus(dte + temp_b[0]);
      float se = sigm((dse + score_b[0]) / taue);
      float jac = fminf(fabsf(se - sc01) * (1.0f / FD_EPS_F), 10.f) * 0.1f;
      out[o_jac + grow] = jac;
    }
  }
}

// ---------------------------------------------------------------------------
// Launch
// ---------------------------------------------------------------------------
extern "C" void kernel_launch(void* const* d_in, const int* in_sizes, int n_in,
                              void* d_out, int out_size, void* d_ws,
                              size_t ws_size, hipStream_t stream) {
  (void)n_in; (void)out_size; (void)ws_size;
  const float* x       = (const float*)d_in[0];
  const float* noise   = (const float*)d_in[1];
  const float* mask    = (const float*)d_in[2];
  const float* proj_w  = (const float*)d_in[3];
  const float* proj_b  = (const float*)d_in[4];
  const float* proj_nw = (const float*)d_in[5];
  const float* l_w_ih  = (const float*)d_in[6];
  const float* l_w_hh  = (const float*)d_in[7];
  const float* l_b_ih  = (const float*)d_in[8];
  const float* l_b_hh  = (const float*)d_in[9];
  const float* l_nw    = (const float*)d_in[10];
  const float* h_w_ih  = (const float*)d_in[11];
  const float* h_w_hh  = (const float*)d_in[12];
  const float* h_b_ih  = (const float*)d_in[13];
  const float* h_b_hh  = (const float*)d_in[14];
  const float* h_nw    = (const float*)d_in[15];
  const float* fin_nw  = (const float*)d_in[16];
  const float* score_w = (const float*)d_in[17];
  const float* score_b = (const float*)d_in[18];
  const float* temp_w  = (const float*)d_in[19];
  const float* temp_b  = (const float*)d_in[20];
  const float* logvar_w = (const float*)d_in[21];
  const float* logvar_b = (const float*)d_in[22];
  const float* aux3_w  = (const float*)d_in[23];
  const float* aux3_b  = (const float*)d_in[24];
  const float* dis_w   = (const float*)d_in[25];
  const float* dis_b   = (const float*)d_in[26];
  const float* cons_w  = (const float*)d_in[27];
  const float* cons_b  = (const float*)d_in[28];
  const float* ood_w   = (const float*)d_in[29];
  const float* ood_b   = (const float*)d_in[30];
  const float* recon_w = (const float*)d_in[31];
  const float* recon_b = (const float*)d_in[32];
  float* out = (float*)d_out;
  int B = in_sizes[0] / IN_DIM;
  size_t Bz = (size_t)B;

  uint8_t* ws = (uint8_t*)d_ws;
  size_t off = 0;
  auto take = [&](size_t bytes) -> void* {
    void* p = ws + off;
    off += (bytes + 255) & ~(size_t)255;
    return p;
  };
  __bf16* w_proj = (__bf16*)take((size_t)IN_DIM * HD * 2);
  __bf16* w_lih  = (__bf16*)take((size_t)(2 * HD) * (3 * LD) * 2);
  __bf16* w_lhh  = (__bf16*)take((size_t)LD * (3 * LD) * 2);
  __bf16* w_hih  = (__bf16*)take((size_t)(LD + HD) * (3 * HD) * 2);
  __bf16* w_hhh  = (__bf16*)take((size_t)HD * (3 * HD) * 2);
  __bf16* w_rec  = (__bf16*)take((size_t)HD * HD * 2);
  float*  invn   = (float*)take(Bz * 4);
  __bf16* xt     = (__bf16*)take(Bz * HD * 2);
  __bf16* xte    = (__bf16*)take(Bz * HD * 2);
  __bf16* zhh    = (__bf16*)take(Bz * HD * 2);
  __bf16* zhe    = (__bf16*)take(Bz * HD * 2);

  auto sw = [&](const float* W, __bf16* dst, int K, int N) {
    int total = (K >> 5) * (N >> 4) * 32;
    swizzle_w<<<(total + 255) / 256, 256, 0, stream>>>(W, dst, K, N);
  };
  sw(proj_w, w_proj, IN_DIM, HD);
  sw(l_w_ih, w_lih, 2 * HD, 3 * LD);
  sw(l_w_hh, w_lhh, LD, 3 * LD);
  sw(h_w_ih, w_hih, LD + HD, 3 * HD);
  sw(h_w_hh, w_hhh, HD, 3 * HD);
  sw(recon_w, w_rec, HD, HD);

  rownorm_kernel<<<(B + 7) / 8, BLOCK, 0, stream>>>(noise, invn, B);

  int nb = B / BM;
  proj_kernel<false><<<nb, BLOCK, 0, stream>>>(x, noise, invn, w_proj, proj_b,
                                               proj_nw, xt);
  proj_kernel<true><<<nb, BLOCK, 0, stream>>>(x, noise, invn, w_proj, proj_b,
                                              proj_nw, xte);

  rollout_kernel<<<nb, BLOCK, 0, stream>>>(
      xt, w_lih, w_lhh, w_hih, w_hhh, l_b_ih, l_b_hh, l_nw, h_b_ih, h_b_hh,
      h_nw, fin_nw, zhh, out + 525 * Bz, out + 781 * Bz, out + 524 * Bz, 1);
  rollout_kernel<<<nb, BLOCK, 0, stream>>>(
      xte, w_lih, w_lhh, w_hih, w_hhh, l_b_ih, l_b_hh, l_nw, h_b_ih, h_b_hh,
      h_nw, fin_nw, zhe, nullptr, nullptr, nullptr, 0);

  heads_kernel<<<nb, BLOCK, 0, stream>>>(
      zhh, zhe, xt, mask, w_rec, recon_b, score_w, score_b, temp_w, temp_b,
      logvar_w, logvar_b, aux3_w, aux3_b, dis_w, dis_b, cons_w, cons_b, ood_w,
      ood_b, out, B);
}
